// QAttention_64467459113268
// MI455X (gfx1250) — matrix operand order repaired and checked by reading, not compiled, not run on hardware
//
#include <hip/hip_runtime.h>
#include <hip/hip_bf16.h>

typedef __attribute__((ext_vector_type(16))) __bf16 v16bf;
typedef __attribute__((ext_vector_type(8)))  __bf16 bf16x8;
typedef __attribute__((ext_vector_type(8)))  float  v8f;

#define BATCH  8
#define SEQ    2048
#define CDIM   768
#define HEADS  12
#define HDIM   64
#define M_ROWS (BATCH*SEQ)   // 16384
#define QKV_N  (3*CDIM)      // 2304
#define KCH    64            // flash attention key-chunk
#define NCH    (SEQ/KCH)     // 32 chunks

#define WMMA_BF16(A, B, C) \
    __builtin_amdgcn_wmma_f32_16x16x32_bf16(false, (A), false, (B), (short)0, (C), false, false)

// ---------- helpers ----------

static __device__ __forceinline__ __bf16 f2bf(float f) {
    unsigned u = __builtin_bit_cast(unsigned, f);
    unsigned r = (u + 0x7FFFu + ((u >> 16) & 1u)) >> 16;
    unsigned short s = (unsigned short)r;
    return __builtin_bit_cast(__bf16, s);
}

// Two aligned 8-element bf16 runs -> one 16-element A/B fragment
static __device__ __forceinline__ v16bf ld2x8(const __bf16* p0, const __bf16* p1) {
    bf16x8 a = *(const bf16x8*)p0;
    bf16x8 b = *(const bf16x8*)p1;
    v16bf r;
#pragma unroll
    for (int i = 0; i < 8; ++i) { r[i] = a[i]; r[i + 8] = b[i]; }
    return r;
}
static __device__ __forceinline__ v16bf ld16(const __bf16* p) {
    return ld2x8(p, p + 8);
}

// gfx1250 async global->LDS copy of 16 bytes per lane
static __device__ __forceinline__ void async_b128(unsigned lds_addr, const void* gaddr) {
    asm volatile("global_load_async_to_lds_b128 %0, %1, off"
                 :: "v"(lds_addr), "v"(gaddr) : "memory");
}
static __device__ __forceinline__ void wait_async0() {
    asm volatile("s_wait_asynccnt 0x0" ::: "memory");
}

// ---------- kernels ----------

__global__ void cast_f32_bf16(const float* __restrict__ src,
                              __bf16* __restrict__ dst, int n) {
    int i = blockIdx.x * blockDim.x + threadIdx.x;
    if (i < n) dst[i] = f2bf(src[i]);
}

// QKV = X @ Wqkv^T ; 32x64 register tile per wave (2 A-frags x 4 B-frags, 8 WMMA/k-step).
// Scatter into q[B,H,N,D] (pre-scaled), k[B,H,N,D], vT[B,H,D,N].
__global__ void __launch_bounds__(256)
qkv_gemm(const __bf16* __restrict__ xb,
         const __bf16* __restrict__ wb,
         __bf16* __restrict__ qb,
         __bf16* __restrict__ kb,
         __bf16* __restrict__ vtb) {
    const int lane = threadIdx.x & 31;
    const int wid  = threadIdx.x >> 5;
    const int NT   = QKV_N / 64;                 // 36
    const int t    = blockIdx.x * 8 + wid;
    if (t >= (M_ROWS / 32) * NT) return;
    const int tm = t / NT, tn = t % NT;
    const int ml = lane & 15, half = lane >> 4;

    const __bf16* a0p = xb + (size_t)(tm * 32 + ml) * CDIM;
    const __bf16* a1p = a0p + (size_t)16 * CDIM;
    const __bf16* bp  = wb + (size_t)(tn * 64 + ml) * CDIM;

    v8f acc[2][4];
#pragma unroll
    for (int i = 0; i < 2; ++i)
#pragma unroll
        for (int j = 0; j < 4; ++j) acc[i][j] = (v8f){};

    for (int kk = 0; kk < CDIM; kk += 32) {
        v16bf a0 = ld2x8(a0p + kk + half * 8, a0p + kk + 16 + half * 8);
        v16bf a1 = ld2x8(a1p + kk + half * 8, a1p + kk + 16 + half * 8);
        __builtin_prefetch(a0p + kk + 128, 0, 3);
        __builtin_prefetch(a1p + kk + 128, 0, 3);
#pragma unroll
        for (int j = 0; j < 4; ++j) {
            v16bf b = ld2x8(bp + (size_t)j * 16 * CDIM + kk + half * 8, bp + (size_t)j * 16 * CDIM + kk + 16 + half * 8);
            acc[0][j] = WMMA_BF16(a0, b, acc[0][j]);
            acc[1][j] = WMMA_BF16(a1, b, acc[1][j]);
        }
    }

    const int cb    = tn * 64;
    const int three = cb / CDIM;       // 64 | 768 -> single (three,h) per wave tile
    const int h     = (cb % CDIM) / HDIM;

#pragma unroll
    for (int i = 0; i < 2; ++i) {
#pragma unroll
        for (int j = 0; j < 4; ++j) {
            const int d = j * 16 + ml;
#pragma unroll
            for (int r = 0; r < 8; ++r) {
                const int m  = tm * 32 + i * 16 + r + 8 * half;
                const int bi = m / SEQ;
                const int n  = m % SEQ;
                const size_t hb = (size_t)(bi * HEADS + h);
                const float v = acc[i][j][r];
                if (three == 0) {
                    qb[(hb * SEQ + n) * HDIM + d] = f2bf(v * 0.125f);  // D^-0.5
                } else if (three == 1) {
                    kb[(hb * SEQ + n) * HDIM + d] = f2bf(v);
                } else {
                    vtb[(hb * HDIM + d) * SEQ + n] = f2bf(v);          // V^T
                }
            }
        }
    }
}

// Flash attention: 4-wave block, 64 queries of one (b,h). 64-key chunks of K/V are
// double-buffered in LDS via gfx1250 async global->LDS DMA, overlapped with compute.
__global__ void __launch_bounds__(128)
flash_attn(const __bf16* __restrict__ qb,
           const __bf16* __restrict__ kb,
           const __bf16* __restrict__ vtb,
           __bf16* __restrict__ ob) {
    __shared__ __align__(16) __bf16 kT[2][KCH * HDIM];   // [key][d]   2 x 8KB
    __shared__ __align__(16) __bf16 vT[2][HDIM * KCH];   // [d][key]   2 x 8KB
    __shared__ __align__(16) __bf16 pl[4][16 * KCH];     // per-wave P transpose

    const int tid  = threadIdx.x;
    const int lane = tid & 31;
    const int wid  = tid >> 5;
    const int ml   = lane & 15, half = lane >> 4;

    const int id = blockIdx.x;
    const int qt = id & 31;                       // SEQ/64 = 32
    const int h  = (id >> 5) % HEADS;
    const int bi = id / (32 * HEADS);

    const size_t hb = (size_t)(bi * HEADS + h);
    const __bf16* qbase = qb  + hb * SEQ  * HDIM;
    const __bf16* kbase = kb  + hb * SEQ  * HDIM;
    const __bf16* vbase = vtb + hb * HDIM * SEQ;

    const int q0 = qt * 64 + wid * 16;            // this wave's 16 queries

    // Q fragments resident in VGPRs: two 16x32 A-frags
    v16bf qa[2];
#pragma unroll
    for (int s = 0; s < 2; ++s) {
        const __bf16* p = qbase + (size_t)(q0 + ml) * HDIM + s * 32 + half * 8;
        qa[s] = ld2x8(p, p + 16);
    }

    v8f o0 = {}, o1 = {}, o2 = {}, o3 = {};
    float mrow[8], lrow[8];
#pragma unroll
    for (int r = 0; r < 8; ++r) { mrow[r] = -1e30f; lrow[r] = 0.f; }

    unsigned kLds[2], vLds[2];
    kLds[0] = (unsigned)(size_t)(&kT[0][0]);
    kLds[1] = (unsigned)(size_t)(&kT[1][0]);
    vLds[0] = (unsigned)(size_t)(&vT[0][0]);
    vLds[1] = (unsigned)(size_t)(&vT[1][0]);

    // Cooperative async stage of one 64-key chunk: K 8KB contiguous, V^T 64 rows x 128B
    auto stage = [&](int buf, int k0s) {
        const char* ksrc = (const char*)(kbase + (size_t)k0s * HDIM);
#pragma unroll
        for (int it = 0; it < 4; ++it) {
            const int u = tid + it * 128;                       // 0..511 (16B units)
            async_b128(kLds[buf] + (unsigned)(u * 16), ksrc + (size_t)u * 16);
            const int d = u >> 3, part = u & 7;
            async_b128(vLds[buf] + (unsigned)(d * 128 + part * 16),
                       (const char*)(vbase + (size_t)d * SEQ + k0s) + part * 16);
        }
    };

    stage(0, 0);
    wait_async0();
    __syncthreads();

    for (int c = 0; c < NCH; ++c) {
        const int cur = c & 1;
        if (c + 1 < NCH) stage(cur ^ 1, (c + 1) * KCH);   // DMA next chunk under compute

        const __bf16* kcur = &kT[cur][0];
        const __bf16* vcur = &vT[cur][0];

        // ---- S = Q K^T : four 16x16 key tiles, K-dim D=64 (2 steps) ----
        v8f sv[4];
#pragma unroll
        for (int kt = 0; kt < 4; ++kt) sv[kt] = (v8f){};
#pragma unroll
        for (int s = 0; s < 2; ++s) {
#pragma unroll
            for (int kt = 0; kt < 4; ++kt) {
                v16bf b = ld2x8(&kcur[(size_t)(kt * 16 + ml) * HDIM + s * 32 + half * 8], &kcur[(size_t)(kt * 16 + ml) * HDIM + s * 32 + 16 + half * 8]);
                sv[kt] = WMMA_BF16(qa[s], b, sv[kt]);
            }
        }

        // ---- online softmax over 64 keys ----
#pragma unroll
        for (int r = 0; r < 8; ++r) {
            float mx = fmaxf(fmaxf(sv[0][r], sv[1][r]), fmaxf(sv[2][r], sv[3][r]));
#pragma unroll
            for (int msk = 8; msk >= 1; msk >>= 1)
                mx = fmaxf(mx, __shfl_xor(mx, msk, 32));
            const float nm   = fmaxf(mrow[r], mx);
            const float corr = __expf(mrow[r] - nm);
            mrow[r] = nm;
            lrow[r] *= corr;
            o0[r] *= corr; o1[r] *= corr; o2[r] *= corr; o3[r] *= corr;
            float ps = 0.f;
#pragma unroll
            for (int kt = 0; kt < 4; ++kt) {
                const float e = __expf(sv[kt][r] - nm);
                sv[kt][r] = e;
                ps += e;
            }
#pragma unroll
            for (int msk = 8; msk >= 1; msk >>= 1)
                ps += __shfl_xor(ps, msk, 32);
            lrow[r] += ps;
        }

        // ---- transpose P (16x64) through per-wave LDS slice (wave-ordered) ----
        __bf16* pw = pl[wid];
#pragma unroll
        for (int kt = 0; kt < 4; ++kt) {
#pragma unroll
            for (int r = 0; r < 8; ++r) {
                const int rr = r + 8 * half;
                pw[rr * KCH + kt * 16 + ml] = f2bf(sv[kt][r]);
            }
        }
        const v16bf pa0 = ld16(&pw[ml * KCH + half * 16]);
        const v16bf pa1 = ld16(&pw[ml * KCH + 32 + half * 16]);

        // ---- O += P @ V : 4 d-tiles x 2 key-halves, B-frags from LDS ----
        {
            v16bf b;
            b = ld16(&vcur[(size_t)(0 * 16 + ml) * KCH + half * 16]);
            o0 = WMMA_BF16(pa0, b, o0);
            b = ld16(&vcur[(size_t)(0 * 16 + ml) * KCH + 32 + half * 16]);
            o0 = WMMA_BF16(pa1, b, o0);
            b = ld16(&vcur[(size_t)(1 * 16 + ml) * KCH + half * 16]);
            o1 = WMMA_BF16(pa0, b, o1);
            b = ld16(&vcur[(size_t)(1 * 16 + ml) * KCH + 32 + half * 16]);
            o1 = WMMA_BF16(pa1, b, o1);
            b = ld16(&vcur[(size_t)(2 * 16 + ml) * KCH + half * 16]);
            o2 = WMMA_BF16(pa0, b, o2);
            b = ld16(&vcur[(size_t)(2 * 16 + ml) * KCH + 32 + half * 16]);
            o2 = WMMA_BF16(pa1, b, o2);
            b = ld16(&vcur[(size_t)(3 * 16 + ml) * KCH + half * 16]);
            o3 = WMMA_BF16(pa0, b, o3);
            b = ld16(&vcur[(size_t)(3 * 16 + ml) * KCH + 32 + half * 16]);
            o3 = WMMA_BF16(pa1, b, o3);
        }

        wait_async0();        // next chunk's DMA done (ran under the compute above)
        __syncthreads();      // all waves done reading cur; next buffer visible
    }

    // ---- normalize, store into [B, N, C] bf16 buffer for projection ----
#pragma unroll
    for (int r = 0; r < 8; ++r) {
        const float inv = 1.0f / lrow[r];
        const int   n   = q0 + r + 8 * half;
        const size_t ro = ((size_t)bi * SEQ + n) * CDIM + h * HDIM;
        ob[ro + 0 * 16 + ml] = f2bf(o0[r] * inv);
        ob[ro + 1 * 16 + ml] = f2bf(o1[r] * inv);
        ob[ro + 2 * 16 + ml] = f2bf(o2[r] * inv);
        ob[ro + 3 * 16 + ml] = f2bf(o3[r] * inv);
    }
}

// out = attn @ Wproj^T + b ; 32x64 tile per wave, fp32 store
__global__ void __launch_bounds__(256)
proj_gemm(const __bf16* __restrict__ ab,
          const __bf16* __restrict__ wb,
          const float* __restrict__ bias,
          float* __restrict__ out) {
    const int lane = threadIdx.x & 31;
    const int wid  = threadIdx.x >> 5;
    const int NT   = CDIM / 64;                  // 12
    const int t    = blockIdx.x * 8 + wid;
    if (t >= (M_ROWS / 32) * NT) return;
    const int tm = t / NT, tn = t % NT;
    const int ml = lane & 15, half = lane >> 4;

    const __bf16* a0p = ab + (size_t)(tm * 32 + ml) * CDIM;
    const __bf16* a1p = a0p + (size_t)16 * CDIM;
    const __bf16* bp  = wb + (size_t)(tn * 64 + ml) * CDIM;

    v8f acc[2][4];
#pragma unroll
    for (int i = 0; i < 2; ++i)
#pragma unroll
        for (int j = 0; j < 4; ++j) acc[i][j] = (v8f){};

    for (int kk = 0; kk < CDIM; kk += 32) {
        v16bf a0 = ld2x8(a0p + kk + half * 8, a0p + kk + 16 + half * 8);
        v16bf a1 = ld2x8(a1p + kk + half * 8, a1p + kk + 16 + half * 8);
        __builtin_prefetch(a0p + kk + 128, 0, 3);
        __builtin_prefetch(a1p + kk + 128, 0, 3);
#pragma unroll
        for (int j = 0; j < 4; ++j) {
            v16bf b = ld2x8(bp + (size_t)j * 16 * CDIM + kk + half * 8, bp + (size_t)j * 16 * CDIM + kk + 16 + half * 8);
            acc[0][j] = WMMA_BF16(a0, b, acc[0][j]);
            acc[1][j] = WMMA_BF16(a1, b, acc[1][j]);
        }
    }

#pragma unroll
    for (int j = 0; j < 4; ++j) {
        const int c  = tn * 64 + j * 16 + ml;
        const float bv = bias[c];
#pragma unroll
        for (int i = 0; i < 2; ++i) {
#pragma unroll
            for (int r = 0; r < 8; ++r) {
                const int m = tm * 32 + i * 16 + r + 8 * half;
                out[(size_t)m * CDIM + c] = acc[i][j][r] + bv;
            }
        }
    }
}

// ---------- launch ----------

extern "C" void kernel_launch(void* const* d_in, const int* in_sizes, int n_in,
                              void* d_out, int out_size, void* d_ws, size_t ws_size,
                              hipStream_t stream) {
    (void)in_sizes; (void)n_in; (void)out_size; (void)ws_size;

    const float* x      = (const float*)d_in[0];
    const float* w_qkv  = (const float*)d_in[1];
    const float* w_proj = (const float*)d_in[2];
    const float* b_proj = (const float*)d_in[3];
    float*       out    = (float*)d_out;

    const size_t nx = (size_t)M_ROWS * CDIM;          // 12,582,912 elems
    __bf16* xb     = (__bf16*)d_ws;
    __bf16* wqkvb  = xb    + nx;
    __bf16* wprojb = wqkvb + (size_t)QKV_N * CDIM;
    __bf16* qb     = wprojb + (size_t)CDIM * CDIM;
    __bf16* kb     = qb  + nx;
    __bf16* vtb    = kb  + nx;
    __bf16* attnb  = vtb + nx;                        // total ~130 MB of ws

    const int nx_i = (int)nx;
    cast_f32_bf16<<<(nx_i + 255) / 256, 256, 0, stream>>>(x, xb, nx_i);
    cast_f32_bf16<<<(QKV_N * CDIM + 255) / 256, 256, 0, stream>>>(w_qkv, wqkvb, QKV_N * CDIM);
    cast_f32_bf16<<<(CDIM * CDIM + 255) / 256, 256, 0, stream>>>(w_proj, wprojb, CDIM * CDIM);

    // (16384/32) * (2304/64) = 18432 wave tiles, 8 waves/block
    qkv_gemm<<<18432 / 8, 256, 0, stream>>>(xb, wqkvb, qb, kb, vtb);

    // B * H * (SEQ/64) = 3072 blocks of 4 waves
    flash_attn<<<BATCH * HEADS * (SEQ / 64), 128, 0, stream>>>(qb, kb, vtb, attnb);

    // (16384/32) * (768/64) = 6144 wave tiles, 8 waves/block
    proj_gemm<<<6144 / 8, 256, 0, stream>>>(attnb, wprojb, b_proj, out);
}
